// DeformableTransformerDecoderLayer_59261958750912
// MI455X (gfx1250) — compile-verified, matrix operand order checked by reading, not run on hardware
//
#include <hip/hip_runtime.h>
#include <hip/hip_bf16.h>

// ---------------- problem constants ----------------
#define B_    16
#define LQ_   900
#define D_    256
#define NH_   8
#define HD_   32
#define NL_   4
#define NP_   4
#define DFFN_ 2048
#define LIN_  21760
#define LQP_  928          // LQ padded to multiple of 32 (alignment for V^T frags)
#define QT_   57           // ceil(900/16)

typedef __attribute__((ext_vector_type(16))) __bf16 bfx16;
typedef __attribute__((ext_vector_type(8)))  float  fltx8;
typedef __attribute__((ext_vector_type(4)))  float  fx4;

// WMMA wrapper: D = A(16x32 bf16) * B(32x16 bf16) + C(16x16 f32)
__device__ __forceinline__ fltx8 wmma_bf16(bfx16 a, bfx16 b, fltx8 c) {
  return __builtin_amdgcn_wmma_f32_16x16x32_bf16(false, a, false, b, (short)0, c,
                                                 false, false);
}

// Load 16 f32 (two 8-element contiguous chunks), scale, convert to bf16 frag.
// Per-lane K layout for 16-bit A (and mirrored B) operands:
//   elems 0..7  -> K = base0 + i        (base0 = 8*(lane>>4))
//   elems 8..15 -> K = 16 + base0 + i-8
__device__ __forceinline__ bfx16 ldcvt(const float* p0, const float* p1, float sc) {
  fx4 a = *(const fx4*)p0;
  fx4 b = *(const fx4*)(p0 + 4);
  fx4 c = *(const fx4*)p1;
  fx4 d = *(const fx4*)(p1 + 4);
  bfx16 r;
  r[0]=(__bf16)(a[0]*sc);  r[1]=(__bf16)(a[1]*sc);  r[2]=(__bf16)(a[2]*sc);  r[3]=(__bf16)(a[3]*sc);
  r[4]=(__bf16)(b[0]*sc);  r[5]=(__bf16)(b[1]*sc);  r[6]=(__bf16)(b[2]*sc);  r[7]=(__bf16)(b[3]*sc);
  r[8]=(__bf16)(c[0]*sc);  r[9]=(__bf16)(c[1]*sc);  r[10]=(__bf16)(c[2]*sc); r[11]=(__bf16)(c[3]*sc);
  r[12]=(__bf16)(d[0]*sc); r[13]=(__bf16)(d[1]*sc); r[14]=(__bf16)(d[2]*sc); r[15]=(__bf16)(d[3]*sc);
  return r;
}

__device__ __forceinline__ fltx8 splat8(float x) {
  fltx8 t;
#pragma unroll
  for (int i = 0; i < 8; ++i) t[i] = x;
  return t;
}

// low 32 bits of a flat pointer to LDS == wave-relative LDS byte offset
__device__ __forceinline__ unsigned lds_off(const void* p) {
  return (unsigned)((uintptr_t)p & 0xffffffffu);
}

// ---------------- generic GEMM: C[M,N] = A[M,K] * W[N,K]^T + bias ----------------
// 8 waves/block -> 128 rows; each wave covers NT*16 output columns (block.y tiles N).
// Per K-step the W tile (NT*16 x 32 f32) is staged in LDS with CDNA5 async
// global->LDS b128 copies (ASYNCcnt), then read back as bf16 WMMA B-fragments.
// TSTORE: C stored per-batch transposed: C[b*N*ldT + n*ldT + (m % rpb)]  (for V^T).
template <bool RELU, bool TSTORE, int NT>
__global__ __launch_bounds__(256) void gemm_kernel(
    const float* __restrict__ A, const float* __restrict__ W,
    const float* __restrict__ bias, float* __restrict__ C,
    int M, int N, int K, int rpb, int ldT) {
  __shared__ float wt[NT * 16][36];            // 144B row stride: 16B aligned, bank-spread
  const int t    = threadIdx.x;
  const int wave = t >> 5;
  const int lane = t & 31;
  const int r    = lane & 15;
  const int hlf  = lane >> 4;
  const int b0   = hlf * 8;
  const int m0   = (blockIdx.x * 8 + wave) * 16;
  const int n0   = blockIdx.y * (NT * 16);

  int arow_i = m0 + r;                          // clamp tail rows (stores are masked)
  if (arow_i > M - 1) arow_i = M - 1;
  const float* arow = A + (size_t)arow_i * K;

  fltx8 acc[NT];
#pragma unroll
  for (int j = 0; j < NT; ++j) acc[j] = splat8(bias ? bias[n0 + 16 * j + r] : 0.f);

  for (int k = 0; k < K; k += 32) {
    // ---- async-stage W[n0 .. n0+NT*16, k .. k+31] into LDS (b128 chunks) ----
#pragma unroll
    for (int i = 0; i < NT / 2; ++i) {
      const int c   = i * 256 + t;              // chunk id: NT*16 rows * 8 chunks
      const int row = c >> 3;
      const int sub = c & 7;
      const unsigned long long ga =
          (unsigned long long)(uintptr_t)(W + (size_t)(n0 + row) * K + k + sub * 4);
      const unsigned lo = lds_off(&wt[row][sub * 4]);
      asm volatile("global_load_async_to_lds_b128 %0, %1, off"
                   :: "v"(lo), "v"(ga) : "memory");
    }
    asm volatile("s_wait_asynccnt 0x0" ::: "memory");
    __syncthreads();

    if (k + 32 < K) __builtin_prefetch(arow + k + 32, 0, 3);
    bfx16 af = ldcvt(arow + k + b0, arow + k + 16 + b0, 1.f);
#pragma unroll
    for (int j = 0; j < NT; ++j) {
      const float* wr = &wt[16 * j + r][0];
      bfx16 bf = ldcvt(wr + b0, wr + 16 + b0, 1.f);
      acc[j] = wmma_bf16(af, bf, acc[j]);
    }
    __syncthreads();                            // before next iteration overwrites wt
  }

#pragma unroll
  for (int j = 0; j < NT; ++j) {
    const int col = n0 + 16 * j + r;
#pragma unroll
    for (int rr = 0; rr < 8; ++rr) {
      const int row = m0 + rr + 8 * hlf;
      if (row < M) {
        float v = acc[j][rr];
        if (RELU) v = v > 0.f ? v : 0.f;
        if (TSTORE) {
          const int bb = row / rpb;
          const int q  = row - bb * rpb;
          C[(size_t)bb * N * ldT + (size_t)col * ldT + q] = v;
        } else {
          C[(size_t)row * N + col] = v;
        }
      }
    }
  }
}

// ---------------- flash attention (transposed tiles) ----------------
// One wave per (b, h, 16-query tile).  S^T = K_blk * Q^T ; O^T += V^T * P^T.
// P^T lands directly in the WMMA B-fragment layout: no cross-lane movement.
__global__ __launch_bounds__(256) void attn_kernel(
    const float* __restrict__ Q, const float* __restrict__ Kf,
    const float* __restrict__ Vt, float* __restrict__ O) {
  const int wid  = (blockIdx.x * blockDim.x + threadIdx.x) >> 5;
  if (wid >= B_ * NH_ * QT_) return;
  const int lane = threadIdx.x & 31;
  const int r    = lane & 15;
  const int hlf  = lane >> 4;
  const int b0   = hlf * 8;
  const int qt   = wid % QT_;
  const int h    = (wid / QT_) % NH_;
  const int b    = wid / (QT_ * NH_);
  const int q0   = qt * 16;

  const float SC = 0.17677669529663687f;  // 1/sqrt(32)

  int qrow = q0 + r; if (qrow > LQ_ - 1) qrow = LQ_ - 1;
  const float* qp = Q + ((size_t)(b * LQ_ + qrow) * D_ + h * HD_);
  const bfx16 qa = ldcvt(qp + b0, qp + 16 + b0, SC);   // doubles as B-frag of Q^T

  float m_i = -1e30f, l_i = 0.f;
  fltx8 o0 = splat8(0.f), o1 = splat8(0.f);   // O^T tiles: hd 0-15 / 16-31

  for (int k0 = 0; k0 < LQ_; k0 += 32) {
    int ka = k0 + r;      if (ka > LQ_ - 1) ka = LQ_ - 1;
    int kb = k0 + 16 + r; if (kb > LQ_ - 1) kb = LQ_ - 1;
    const float* kp0 = Kf + ((size_t)(b * LQ_ + ka) * D_ + h * HD_);
    const float* kp1 = Kf + ((size_t)(b * LQ_ + kb) * D_ + h * HD_);
    bfx16 kfa = ldcvt(kp0 + b0, kp0 + 16 + b0, 1.f);
    bfx16 kfb = ldcvt(kp1 + b0, kp1 + 16 + b0, 1.f);

    fltx8 st0 = splat8(0.f), st1 = splat8(0.f);
    st0 = wmma_bf16(kfa, qa, st0);     // keys k0..k0+15  x queries
    st1 = wmma_bf16(kfb, qa, st1);     // keys k0+16..+31 x queries

    float s0[8], s1[8];
    float mloc = -1e30f;
#pragma unroll
    for (int rr = 0; rr < 8; ++rr) {
      const int key0 = k0 + rr + b0;
      s0[rr] = (key0      < LQ_) ? st0[rr] : -1e30f;
      s1[rr] = (key0 + 16 < LQ_) ? st1[rr] : -1e30f;
      mloc = fmaxf(mloc, fmaxf(s0[rr], s1[rr]));
    }
    mloc = fmaxf(mloc, __shfl_xor(mloc, 16, 32));   // combine key halves (same query col)
    const float mnew  = fmaxf(m_i, mloc);
    const float alpha = __expf(m_i - mnew);
    float lsum = 0.f;
    bfx16 pb;                                        // P^T already in B-frag layout
#pragma unroll
    for (int rr = 0; rr < 8; ++rr) {
      const float p0 = __expf(s0[rr] - mnew);
      const float p1 = __expf(s1[rr] - mnew);
      lsum += p0 + p1;
      pb[rr]     = (__bf16)p0;
      pb[rr + 8] = (__bf16)p1;
    }
    lsum += __shfl_xor(lsum, 16, 32);
    l_i = l_i * alpha + lsum;
    m_i = mnew;
#pragma unroll
    for (int rr = 0; rr < 8; ++rr) { o0[rr] *= alpha; o1[rr] *= alpha; }

    const float* vp0 = Vt + ((size_t)(b * D_ + h * HD_ + r) * LQP_) + k0;
    const float* vp1 = vp0 + (size_t)16 * LQP_;
    bfx16 va0 = ldcvt(vp0 + b0, vp0 + 16 + b0, 1.f);
    bfx16 va1 = ldcvt(vp1 + b0, vp1 + 16 + b0, 1.f);
    o0 = wmma_bf16(va0, pb, o0);
    o1 = wmma_bf16(va1, pb, o1);
  }

  const float inv = 1.f / l_i;
  const int q = q0 + r;
  if (q < LQ_) {
    const size_t base = (size_t)(b * LQ_ + q) * D_ + h * HD_;
#pragma unroll
    for (int rr = 0; rr < 8; ++rr) {
      const int hd = rr + 8 * hlf;
      O[base + hd]      = o0[rr] * inv;
      O[base + 16 + hd] = o1[rr] * inv;
    }
  }
}

// ---------------- LayerNorm: Y = LN(X + R) * g + b (wave per row, D=256) ----------------
__global__ __launch_bounds__(256) void ln_kernel(
    const float* __restrict__ X, const float* __restrict__ R,
    const float* __restrict__ g, const float* __restrict__ be,
    float* __restrict__ Y, int M) {
  const int wid  = (blockIdx.x * blockDim.x + threadIdx.x) >> 5;
  const int lane = threadIdx.x & 31;
  if (wid >= M) return;
  const float* x = X + (size_t)wid * D_;
  const float* rr = R + (size_t)wid * D_;
  float v[8];
  float s = 0.f;
#pragma unroll
  for (int i = 0; i < 8; ++i) { v[i] = x[lane + 32 * i] + rr[lane + 32 * i]; s += v[i]; }
#pragma unroll
  for (int m = 1; m < 32; m <<= 1) s += __shfl_xor(s, m, 32);
  const float mean = s * (1.f / D_);
  float vs = 0.f;
#pragma unroll
  for (int i = 0; i < 8; ++i) { const float d = v[i] - mean; vs += d * d; }
#pragma unroll
  for (int m = 1; m < 32; m <<= 1) vs += __shfl_xor(vs, m, 32);
  const float rstd = rsqrtf(vs * (1.f / D_) + 1e-5f);
#pragma unroll
  for (int i = 0; i < 8; ++i) {
    const int c = lane + 32 * i;
    Y[(size_t)wid * D_ + c] = (v[i] - mean) * rstd * g[c] + be[c];
  }
}

// ---------------- softmax over 16 attention weights per (b,q,h) ----------------
__global__ __launch_bounds__(256) void softmax16_kernel(
    const float* __restrict__ X, float* __restrict__ Y, int rows) {
  const int t = blockIdx.x * blockDim.x + threadIdx.x;
  if (t >= rows) return;
  const int m = t >> 3, h = t & 7;
  const float* x = X + (size_t)m * 128 + h * 16;
  float* y = Y + (size_t)m * 128 + h * 16;
  float mx = -1e30f;
#pragma unroll
  for (int i = 0; i < 16; ++i) mx = fmaxf(mx, x[i]);
  float e[16], s = 0.f;
#pragma unroll
  for (int i = 0; i < 16; ++i) { e[i] = __expf(x[i] - mx); s += e[i]; }
  const float inv = 1.f / s;
#pragma unroll
  for (int i = 0; i < 16; ++i) y[i] = e[i] * inv;
}

// ---------------- multi-scale deformable sampling (wave per (b,q,h), lane = channel) ----
__global__ __launch_bounds__(256) void msdeform_kernel(
    const float* __restrict__ value, const float* __restrict__ refp,
    const float* __restrict__ off, const float* __restrict__ attw,
    float* __restrict__ out) {
  const int wid  = (blockIdx.x * blockDim.x + threadIdx.x) >> 5;
  const int lane = threadIdx.x & 31;
  const int h = wid & 7;
  const int m = wid >> 3;         // b*LQ + q
  const int b = m / LQ_;
  const int WLS[4] = {128, 64, 32, 16};
  const int HLS[4] = {128, 64, 32, 16};
  const int ST[4]  = {0, 16384, 20480, 21504};
  const float* om = off  + (size_t)m * 256 + h * 32;
  const float* am = attw + (size_t)m * 128 + h * 16;
  const float* rp = refp + (size_t)m * (NL_ * 2);
  float acc = 0.f;
#pragma unroll
  for (int l = 0; l < NL_; ++l) {
    const int wl = WLS[l], hl = HLS[l];
    const float wlf = (float)wl, hlf2 = (float)hl;
    const float rx = rp[l * 2 + 0], ry = rp[l * 2 + 1];
    const float* vbase = value + ((size_t)(b * LIN_ + ST[l])) * D_ + h * HD_ + lane;
#pragma unroll
    for (int p = 0; p < NP_; ++p) {
      const float gx = rx + om[l * 8 + p * 2 + 0] / wlf;
      const float gy = ry + om[l * 8 + p * 2 + 1] / hlf2;
      const float x = gx * wlf - 0.5f, y = gy * hlf2 - 0.5f;
      const float x0f = floorf(x), y0f = floorf(y);
      const float fx = x - x0f, fy = y - y0f;
      const int x0 = (int)x0f, y0 = (int)y0f;
      float sacc = 0.f;
#pragma unroll
      for (int dy = 0; dy < 2; ++dy) {
#pragma unroll
        for (int dx = 0; dx < 2; ++dx) {
          const int xi = x0 + dx, yi = y0 + dy;
          const bool valid = (xi >= 0) & (xi < wl) & (yi >= 0) & (yi < hl);
          const int xc = min(max(xi, 0), wl - 1);
          const int yc = min(max(yi, 0), hl - 1);
          const float w = valid ? (dy ? fy : 1.f - fy) * (dx ? fx : 1.f - fx) : 0.f;
          sacc += w * vbase[(size_t)(yc * wl + xc) * D_];
        }
      }
      acc += am[l * 4 + p] * sacc;
    }
  }
  out[(size_t)m * 256 + h * 32 + lane] = acc;
}

// ---------------- elementwise helpers ----------------
__global__ void add_kernel(const float* __restrict__ a, const float* __restrict__ b,
                           float* __restrict__ o, int n) {
  for (int i = blockIdx.x * blockDim.x + threadIdx.x; i < n; i += gridDim.x * blockDim.x)
    o[i] = a[i] + b[i];
}
__global__ void zero_kernel(float* __restrict__ o, int n) {
  for (int i = blockIdx.x * blockDim.x + threadIdx.x; i < n; i += gridDim.x * blockDim.x)
    o[i] = 0.f;
}

// ---------------- host driver ----------------
extern "C" void kernel_launch(void* const* d_in, const int* in_sizes, int n_in,
                              void* d_out, int out_size, void* d_ws, size_t ws_size,
                              hipStream_t stream) {
  const float* queries      = (const float*)d_in[0];
  const float* features     = (const float*)d_in[1];
  const float* refp         = (const float*)d_in[2];
  const float* q_pos        = (const float*)d_in[3];
  const float* in_proj_w    = (const float*)d_in[4];
  const float* in_proj_b    = (const float*)d_in[5];
  const float* out_proj_w   = (const float*)d_in[6];
  const float* out_proj_b   = (const float*)d_in[7];
  const float* value_proj_w = (const float*)d_in[8];
  const float* value_proj_b = (const float*)d_in[9];
  const float* samp_w       = (const float*)d_in[10];
  const float* samp_b       = (const float*)d_in[11];
  const float* attw_w       = (const float*)d_in[12];
  const float* attw_b       = (const float*)d_in[13];
  const float* msout_w      = (const float*)d_in[14];
  const float* msout_b      = (const float*)d_in[15];
  const float* ffn_w1       = (const float*)d_in[16];
  const float* ffn_b1       = (const float*)d_in[17];
  const float* ffn_w2       = (const float*)d_in[18];
  const float* ffn_b2       = (const float*)d_in[19];
  const float* ln1_g = (const float*)d_in[20];
  const float* ln1_b = (const float*)d_in[21];
  const float* ln2_g = (const float*)d_in[22];
  const float* ln2_b = (const float*)d_in[23];
  const float* ln3_g = (const float*)d_in[24];
  const float* ln3_b = (const float*)d_in[25];
  (void)in_sizes; (void)n_in; (void)out_size; (void)ws_size;

  const int MQ = B_ * LQ_;    // 14400
  const int MV = B_ * LIN_;   // 348160

  float* ws = (float*)d_ws;
  size_t o = 0;
  auto alloc = [&](size_t nflt) { float* p = ws + o; o += nflt; return p; };
  float* value    = alloc((size_t)MV * D_);          // 89.1M f
  float* qk       = alloc((size_t)MQ * D_);
  float* Qb       = alloc((size_t)MQ * D_);
  float* Kb       = alloc((size_t)MQ * D_);
  float* Vt       = alloc((size_t)B_ * D_ * LQP_);   // transposed V, padded keys
  float* attn_out = alloc((size_t)MQ * D_);
  float* mha_proj = alloc((size_t)MQ * D_);
  float* y1       = alloc((size_t)MQ * D_);
  float* q2       = alloc((size_t)MQ * D_);
  float* offb     = alloc((size_t)MQ * 256);
  float* awl      = alloc((size_t)MQ * 128);
  float* aws      = alloc((size_t)MQ * 128);
  float* msout    = alloc((size_t)MQ * D_);
  float* msproj   = alloc((size_t)MQ * D_);
  float* y2       = alloc((size_t)MQ * D_);
  float* ffnh     = alloc((size_t)MQ * DFFN_);
  float* ffno     = alloc((size_t)MQ * D_);

  const dim3 TB(256);
  const int gM = (MQ + 127) / 128;   // 113

  // qk = queries + q_pos ; zero V^T (pad region must be finite/zero)
  add_kernel<<<dim3(2048), TB, 0, stream>>>(queries, q_pos, qk, MQ * D_);
  zero_kernel<<<dim3(2048), TB, 0, stream>>>(Vt, (int)((size_t)B_ * D_ * LQP_));

  // Q,K,V projections (V stored transposed per batch for the attention kernel)
  gemm_kernel<false, false, 16><<<dim3(gM, 1), TB, 0, stream>>>(qk, in_proj_w, in_proj_b, Qb, MQ, 256, 256, 1, 1);
  gemm_kernel<false, false, 16><<<dim3(gM, 1), TB, 0, stream>>>(qk, in_proj_w + 256 * 256, in_proj_b + 256, Kb, MQ, 256, 256, 1, 1);
  gemm_kernel<false, true , 16><<<dim3(gM, 1), TB, 0, stream>>>(queries, in_proj_w + 512 * 256, in_proj_b + 512, Vt, MQ, 256, 256, LQ_, LQP_);

  // value projection of the multi-scale features (the big GEMM; A streamed once)
  gemm_kernel<false, false, 16><<<dim3(MV / 128, 1), TB, 0, stream>>>(features, value_proj_w, value_proj_b, value, MV, 256, 256, 1, 1);

  // self-attention + out-proj + LN1
  attn_kernel<<<dim3((B_ * NH_ * QT_) / 8), TB, 0, stream>>>(Qb, Kb, Vt, attn_out);
  gemm_kernel<false, false, 16><<<dim3(gM, 1), TB, 0, stream>>>(attn_out, out_proj_w, out_proj_b, mha_proj, MQ, 256, 256, 1, 1);
  ln_kernel<<<dim3(MQ / 8), TB, 0, stream>>>(mha_proj, queries, ln1_g, ln1_b, y1, MQ);

  // deformable branch
  add_kernel<<<dim3(2048), TB, 0, stream>>>(y1, q_pos, q2, MQ * D_);
  gemm_kernel<false, false, 16><<<dim3(gM, 1), TB, 0, stream>>>(q2, samp_w, samp_b, offb, MQ, 256, 256, 1, 1);
  gemm_kernel<false, false, 8><<<dim3(gM, 1), TB, 0, stream>>>(q2, attw_w, attw_b, awl, MQ, 128, 256, 1, 1);
  softmax16_kernel<<<dim3((MQ * NH_ + 255) / 256), TB, 0, stream>>>(awl, aws, MQ * NH_);
  msdeform_kernel<<<dim3((MQ * NH_) / 8), TB, 0, stream>>>(value, refp, offb, aws, msout);
  gemm_kernel<false, false, 16><<<dim3(gM, 1), TB, 0, stream>>>(msout, msout_w, msout_b, msproj, MQ, 256, 256, 1, 1);
  ln_kernel<<<dim3(MQ / 8), TB, 0, stream>>>(msproj, y1, ln2_g, ln2_b, y2, MQ);

  // FFN + LN3 -> output
  gemm_kernel<true,  false, 16><<<dim3(gM, 8), TB, 0, stream>>>(y2, ffn_w1, ffn_b1, ffnh, MQ, DFFN_, 256, 1, 1);
  gemm_kernel<false, false, 16><<<dim3(gM, 1), TB, 0, stream>>>(ffnh, ffn_w2, ffn_b2, ffno, MQ, 256, DFFN_, 1, 1);
  ln_kernel<<<dim3(MQ / 8), TB, 0, stream>>>(ffno, y2, ln3_g, ln3_b, (float*)d_out, MQ);
}
